// GAT_6880537608210
// MI455X (gfx1250) — compile-verified, matrix operand order checked by reading, not run on hardware
//
#include <hip/hip_runtime.h>
#include <hip/hip_bf16.h>

typedef __attribute__((ext_vector_type(16))) _Float16 v16h;
typedef __attribute__((ext_vector_type(8)))  float    v8f;

#define NEG_SLOPE 0.2f

// ---------------------------------------------------------------------------
// B pre-pack: convert B[K,N] f32 into the exact WMMA 16x16x32_f16 B-fragment
// per-lane layout: unit (ntile, kstep, lane) holds 16 halves (32 bytes) so the
// GEMM reads one aligned 32B vector per fragment.
//   k = 16*(v>>2) + 8*hi + 2*(v&3) + h   (within a 32-wide kstep)
// ---------------------------------------------------------------------------
__global__ void gat_pack_bfrag(const float* __restrict__ B, _Float16* __restrict__ F,
                               int K, int N) {
    long long idx = (long long)blockIdx.x * blockDim.x + threadIdx.x;
    if (idx >= (long long)K * N) return;
    int k = (int)(idx / N), n = (int)(idx % N);
    int nt = n >> 4, col = n & 15;
    int ks = k >> 5, kr = k & 31;
    int hi = (kr >> 3) & 1;
    int v  = ((kr >> 4) << 2) | ((kr >> 1) & 3);
    int id2 = (v << 1) | (kr & 1);
    int lane = (hi << 4) | col;
    F[(((size_t)nt * (K >> 5) + ks) * 32 + lane) * 16 + id2] = (_Float16)B[idx];
}

// A fragment: lane's 16 halves are two contiguous 8-float runs of its row.
__device__ __forceinline__ v16h load_a_frag(const float* __restrict__ Arow,
                                            int kt, int hi) {
    const float4* p0 = (const float4*)(Arow + kt + (hi << 3));
    const float4* p1 = (const float4*)(Arow + kt + 16 + (hi << 3));
    float4 x0 = p0[0], x1 = p0[1];
    float4 y0 = p1[0], y1 = p1[1];
    v16h a;
    a[0]  = (_Float16)x0.x; a[1]  = (_Float16)x0.y;
    a[2]  = (_Float16)x0.z; a[3]  = (_Float16)x0.w;
    a[4]  = (_Float16)x1.x; a[5]  = (_Float16)x1.y;
    a[6]  = (_Float16)x1.z; a[7]  = (_Float16)x1.w;
    a[8]  = (_Float16)y0.x; a[9]  = (_Float16)y0.y;
    a[10] = (_Float16)y0.z; a[11] = (_Float16)y0.w;
    a[12] = (_Float16)y1.x; a[13] = (_Float16)y1.y;
    a[14] = (_Float16)y1.z; a[15] = (_Float16)y1.w;
    return a;
}

__device__ __forceinline__ v16h load_b_frag(const _Float16* __restrict__ Bf,
                                            int tile, int lane) {
    return *(const v16h*)(Bf + ((size_t)tile * 32 + lane) * 16);
}

__device__ __forceinline__ void store_c_tile(float* __restrict__ C, const v8f& acc,
                                             int mBase, int col, int hi, int M, int N,
                                             bool full) {
    if (full) {
#pragma unroll
        for (int r = 0; r < 8; ++r)
            C[(size_t)(mBase + r + (hi << 3)) * N + col] = acc[r];
    } else {
#pragma unroll
        for (int r = 0; r < 8; ++r) {
            int cRow = mBase + r + (hi << 3);
            if (cRow < M) C[(size_t)cRow * N + col] = acc[r];
        }
    }
}

// ---------------------------------------------------------------------------
// GEMM variant 1: A-resident (K = KSTEPS*32 fits in regs). One wave per 16-row
// strip keeps all A fragments in VGPRs and sweeps the column tiles.
// ---------------------------------------------------------------------------
template <int KSTEPS>
__global__ void gat_gemm_aresident(const float* __restrict__ A,
                                   const _Float16* __restrict__ Bf,
                                   float* __restrict__ C, int M, int N) {
    const int K = KSTEPS * 32;
    int wave  = blockIdx.x * (blockDim.x >> 5) + (threadIdx.x >> 5);
    int mBase = wave << 4;
    if (mBase >= M) return;
    int lane = threadIdx.x & 31, laneLow = lane & 15, hi = lane >> 4;
    bool full = (mBase + 16 <= M);
    int aRow = mBase + laneLow;
    if (aRow >= M) aRow = M - 1;
    const float* Arow = A + (size_t)aRow * K;

    v16h a[KSTEPS];
#pragma unroll
    for (int ks = 0; ks < KSTEPS; ++ks) a[ks] = load_a_frag(Arow, ks * 32, hi);

    for (int nt = 0; nt < N; nt += 16) {
        v8f acc = {};
#pragma unroll
        for (int ks = 0; ks < KSTEPS; ++ks) {
            v16h b = load_b_frag(Bf, (nt >> 4) * KSTEPS + ks, lane);
            acc = __builtin_amdgcn_wmma_f32_16x16x32_f16(
                false, a[ks], false, b, (short)0, acc, false, false);
        }
        store_c_tile(C, acc, mBase, nt + laneLow, hi, M, N, full);
    }
}

// ---------------------------------------------------------------------------
// GEMM variant 2: multi-accumulator (NT column tiles live in regs), stream K
// once. NT=1 doubles as the generic fallback. Requires N % (NT*16) == 0.
// ---------------------------------------------------------------------------
template <int NT>
__global__ void gat_gemm_multiacc(const float* __restrict__ A,
                                  const _Float16* __restrict__ Bf,
                                  float* __restrict__ C, int M, int K, int N) {
    int wave  = blockIdx.x * (blockDim.x >> 5) + (threadIdx.x >> 5);
    int mBase = wave << 4;
    if (mBase >= M) return;
    int lane = threadIdx.x & 31, laneLow = lane & 15, hi = lane >> 4;
    bool full = (mBase + 16 <= M);
    int aRow = mBase + laneLow;
    if (aRow >= M) aRow = M - 1;
    const float* Arow = A + (size_t)aRow * K;
    const int KS = K >> 5;

    for (int nt0 = 0; nt0 < N; nt0 += NT * 16) {
        v8f acc[NT];
#pragma unroll
        for (int j = 0; j < NT; ++j) acc[j] = (v8f){};
        for (int kt = 0; kt < K; kt += 32) {
            v16h a = load_a_frag(Arow, kt, hi);
#pragma unroll
            for (int j = 0; j < NT; ++j) {
                v16h b = load_b_frag(Bf, ((nt0 >> 4) + j) * KS + (kt >> 5), lane);
                acc[j] = __builtin_amdgcn_wmma_f32_16x16x32_f16(
                    false, a, false, b, (short)0, acc[j], false, false);
            }
        }
#pragma unroll
        for (int j = 0; j < NT; ++j)
            store_c_tile(C, acc[j], mBase, nt0 + j * 16 + laneLow, hi, M, N, full);
    }
}

// ---------------------------------------------------------------------------
// alpha_s / alpha_d: one wave per (node, head), lanes cover the 64 channels.
// ---------------------------------------------------------------------------
__global__ void gat_alpha(const float* __restrict__ h,
                          const float* __restrict__ a_src,
                          const float* __restrict__ a_dst,
                          float* __restrict__ as, float* __restrict__ ad,
                          int Nn, int H) {
    const int C = 64;
    int wave = blockIdx.x * (blockDim.x >> 5) + (threadIdx.x >> 5);
    if (wave >= Nn * H) return;
    int n = wave / H, hh = wave % H;
    int lane = threadIdx.x & 31;
    const float* hp  = h + ((size_t)n * H + hh) * C;
    const float* asp = a_src + hh * C;
    const float* adp = a_dst + hh * C;
    float v0 = hp[lane], v1 = hp[lane + 32];
    float ss = v0 * asp[lane] + v1 * asp[lane + 32];
    float sd = v0 * adp[lane] + v1 * adp[lane + 32];
#pragma unroll
    for (int off = 16; off > 0; off >>= 1) {
        ss += __shfl_down(ss, off, 32);
        sd += __shfl_down(sd, off, 32);
    }
    if (lane == 0) { as[wave] = ss; ad[wave] = sd; }
}

// order-preserving float<->uint for atomic segment-max
__device__ __forceinline__ unsigned fenc(float f) {
    unsigned u = __float_as_uint(f);
    return (u & 0x80000000u) ? ~u : (u | 0x80000000u);
}
__device__ __forceinline__ float fdec(unsigned u) {
    return __uint_as_float((u & 0x80000000u) ? (u & 0x7FFFFFFFu) : ~u);
}

__device__ __forceinline__ void edge_sd(const int* src, const int* dst,
                                        int E, int eid, int& s, int& d) {
    if (eid < E) { s = src[eid]; d = dst[eid]; }
    else         { s = d = eid - E; }            // virtual self-loop
}

__global__ void gat_edge_max(const int* __restrict__ src, const int* __restrict__ dst,
                             int E, int Nn, int H,
                             const float* __restrict__ as, const float* __restrict__ ad,
                             unsigned* __restrict__ menc) {
    long long idx = (long long)blockIdx.x * blockDim.x + threadIdx.x;
    if (idx >= (long long)(E + Nn) * H) return;
    int eid = (int)(idx / H), hh = (int)(idx % H);
    int s, d; edge_sd(src, dst, E, eid, s, d);
    float e = as[s * H + hh] + ad[d * H + hh];
    e = e > 0.f ? e : NEG_SLOPE * e;
    atomicMax(&menc[d * H + hh], fenc(e));
}

__global__ void gat_edge_sum(const int* __restrict__ src, const int* __restrict__ dst,
                             int E, int Nn, int H,
                             const float* __restrict__ as, const float* __restrict__ ad,
                             const unsigned* __restrict__ menc,
                             float* __restrict__ denom) {
    long long idx = (long long)blockIdx.x * blockDim.x + threadIdx.x;
    if (idx >= (long long)(E + Nn) * H) return;
    int eid = (int)(idx / H), hh = (int)(idx % H);
    int s, d; edge_sd(src, dst, E, eid, s, d);
    float e = as[s * H + hh] + ad[d * H + hh];
    e = e > 0.f ? e : NEG_SLOPE * e;
    atomicAdd(&denom[d * H + hh], __expf(e - fdec(menc[d * H + hh])));
}

// out[dst] += alpha * h[src]; one wave per (edge, head), lanes = channels.
__global__ void gat_edge_scatter(const int* __restrict__ src, const int* __restrict__ dst,
                                 int E, int Nn, int H,
                                 const float* __restrict__ as, const float* __restrict__ ad,
                                 const unsigned* __restrict__ menc,
                                 const float* __restrict__ denom,
                                 const float* __restrict__ hsrc,
                                 float* __restrict__ out) {
    const int C = 64;
    long long wave = (long long)blockIdx.x * (blockDim.x >> 5) + (threadIdx.x >> 5);
    if (wave >= (long long)(E + Nn) * H) return;
    int eid = (int)(wave / H), hh = (int)(wave % H);
    int s, d; edge_sd(src, dst, E, eid, s, d);
    int lane = threadIdx.x & 31;
    float e = as[s * H + hh] + ad[d * H + hh];
    e = e > 0.f ? e : NEG_SLOPE * e;
    float alpha = __expf(e - fdec(menc[d * H + hh])) / denom[d * H + hh];
    const float* hp = hsrc + ((size_t)s * H + hh) * C;
    float*       op = out  + ((size_t)d * H + hh) * C;
    atomicAdd(&op[lane],      hp[lane]      * alpha);
    atomicAdd(&op[lane + 32], hp[lane + 32] * alpha);
}

__global__ void gat_bias_relu(float* __restrict__ h, const float* __restrict__ b,
                              long long total, int D) {
    long long idx = (long long)blockIdx.x * blockDim.x + threadIdx.x;
    if (idx >= total) return;
    float v = h[idx] + b[(int)(idx % D)];
    h[idx] = v > 0.f ? v : 0.f;
}

__global__ void gat_fill0(unsigned* __restrict__ p, long long n) {
    long long idx = (long long)blockIdx.x * blockDim.x + threadIdx.x;
    if (idx < n) p[idx] = 0u;
}

__global__ void gat_pool(const float* __restrict__ h, const int* __restrict__ batch,
                         float* __restrict__ pooled, int Nn) {
    long long idx = (long long)blockIdx.x * blockDim.x + threadIdx.x;
    if (idx >= (long long)Nn * 64) return;
    int n = (int)(idx >> 6), c = (int)(idx & 63);
    atomicAdd(&pooled[batch[n] * 64 + c], h[idx]);
}

__global__ void gat_fc(const float* __restrict__ pooled, const float* __restrict__ W,
                       const float* __restrict__ b, float* __restrict__ out,
                       int G, int Cin, int Cout) {
    int idx = blockIdx.x * blockDim.x + threadIdx.x;
    if (idx >= G * Cout) return;
    int g = idx / Cout, o = idx % Cout;
    float s = b[o];
    for (int c = 0; c < Cin; ++c) s += pooled[g * Cin + c] * W[c * Cout + o];
    out[idx] = s;
}

static inline int cdiv(long long a, long long b) { return (int)((a + b - 1) / b); }

extern "C" void kernel_launch(void* const* d_in, const int* in_sizes, int n_in,
                              void* d_out, int out_size, void* d_ws, size_t ws_size,
                              hipStream_t stream) {
    const float* x      = (const float*)d_in[0];
    const int*   ei     = (const int*)d_in[1];
    const int*   batch  = (const int*)d_in[2];
    const float* W1     = (const float*)d_in[3];
    const float* a_src1 = (const float*)d_in[4];
    const float* a_dst1 = (const float*)d_in[5];
    const float* b1     = (const float*)d_in[6];
    const float* W2     = (const float*)d_in[7];
    const float* a_src2 = (const float*)d_in[8];
    const float* a_dst2 = (const float*)d_in[9];
    const float* b2     = (const float*)d_in[10];
    const float* Wfc    = (const float*)d_in[11];
    const float* bfc    = (const float*)d_in[12];
    float* out = (float*)d_out;

    const int hid   = in_sizes[10];              // 64
    const int H1    = in_sizes[4] / hid;         // 8
    const int D1    = H1 * hid;                  // 512
    const int inDim = in_sizes[3] / D1;          // 128
    const int Nn    = in_sizes[0] / inDim;       // 50000
    const int E     = in_sizes[1] / 2;           // 800000
    const int Cout  = in_sizes[12];              // 10
    const int G     = out_size / Cout;           // 64
    const int* src = ei;
    const int* dst = ei + E;

    // ---- workspace layout (floats); layer-2 buffers alias dead h1lin ----
    size_t NB = (size_t)Nn;
    float* h1lin = (float*)d_ws;                 // NB*D1
    float* h1agg = h1lin + NB * D1;              // NB*D1
    float* as1   = h1agg + NB * D1;              // NB*H1
    float* ad1   = as1 + NB * H1;
    unsigned* m1 = (unsigned*)(ad1 + NB * H1);   // NB*H1
    float* den1  = (float*)(m1 + NB * H1);       // NB*H1
    float* pooled= den1 + NB * H1;               // G*hid
    _Float16* bf1 = (_Float16*)(pooled + (size_t)G * hid);   // inDim*D1 halves
    _Float16* bf2 = bf1 + (size_t)inDim * D1;                // D1*hid halves
    // layer-2 aliases inside h1lin region:
    float* h2lin = h1lin;                        // NB*hid
    float* h2agg = h2lin + NB * hid;             // NB*hid
    float* as2   = h2agg + NB * hid;             // NB
    float* ad2   = as2 + NB;
    unsigned* m2 = (unsigned*)(ad2 + NB);        // NB
    float* den2  = (float*)(m2 + NB);            // NB
    (void)ws_size; (void)n_in;

    const int BT = 256;
    long long edges1 = (long long)(E + Nn) * H1;
    long long edges2 = (long long)(E + Nn);
    int gemmGrid = cdiv((Nn + 15) / 16, BT / 32);

    // ---- pack both B matrices into WMMA fragment layout (one-shot, tiny) ----
    gat_pack_bfrag<<<cdiv((long long)inDim * D1, BT), BT, 0, stream>>>(W1, bf1, inDim, D1);
    gat_pack_bfrag<<<cdiv((long long)D1 * hid, BT), BT, 0, stream>>>(W2, bf2, D1, hid);

    // ---- zero h1agg + as1/ad1/m1/den1 + pooled (contiguous span) ----
    long long z1 = (long long)NB * D1 + 4LL * NB * H1 + (long long)G * hid;
    gat_fill0<<<cdiv(z1, BT), BT, 0, stream>>>((unsigned*)h1agg, z1);

    // ---- layer 1: x @ W1 (A-resident: K=128 fragments live in VGPRs) ----
    if (inDim == 128)
        gat_gemm_aresident<4><<<gemmGrid, BT, 0, stream>>>(x, bf1, h1lin, Nn, D1);
    else
        gat_gemm_multiacc<1><<<gemmGrid, BT, 0, stream>>>(x, bf1, h1lin, Nn, inDim, D1);
    gat_alpha<<<cdiv((long long)Nn * H1, BT / 32), BT, 0, stream>>>(
        h1lin, a_src1, a_dst1, as1, ad1, Nn, H1);
    gat_edge_max<<<cdiv(edges1, BT), BT, 0, stream>>>(src, dst, E, Nn, H1, as1, ad1, m1);
    gat_edge_sum<<<cdiv(edges1, BT), BT, 0, stream>>>(src, dst, E, Nn, H1, as1, ad1, m1, den1);
    gat_edge_scatter<<<cdiv(edges1, BT / 32), BT, 0, stream>>>(
        src, dst, E, Nn, H1, as1, ad1, m1, den1, h1lin, h1agg);
    gat_bias_relu<<<cdiv((long long)NB * D1, BT), BT, 0, stream>>>(
        h1agg, b1, (long long)NB * D1, D1);

    // ---- zero layer-2 scratch (inside now-dead h1lin region) ----
    long long z2 = (long long)NB * hid + 4LL * NB;
    gat_fill0<<<cdiv(z2, BT), BT, 0, stream>>>((unsigned*)h2agg, z2);

    // ---- layer 2: h1 @ W2 (multi-acc: 4 column tiles, stream K once) ----
    if ((hid & 63) == 0)
        gat_gemm_multiacc<4><<<gemmGrid, BT, 0, stream>>>(h1agg, bf2, h2lin, Nn, D1, hid);
    else
        gat_gemm_multiacc<1><<<gemmGrid, BT, 0, stream>>>(h1agg, bf2, h2lin, Nn, D1, hid);
    gat_alpha<<<cdiv((long long)Nn, BT / 32), BT, 0, stream>>>(
        h2lin, a_src2, a_dst2, as2, ad2, Nn, 1);
    gat_edge_max<<<cdiv(edges2, BT), BT, 0, stream>>>(src, dst, E, Nn, 1, as2, ad2, m2);
    gat_edge_sum<<<cdiv(edges2, BT), BT, 0, stream>>>(src, dst, E, Nn, 1, as2, ad2, m2, den2);
    gat_edge_scatter<<<cdiv(edges2, BT / 32), BT, 0, stream>>>(
        src, dst, E, Nn, 1, as2, ad2, m2, den2, h2lin, h2agg);
    gat_bias_relu<<<cdiv((long long)NB * hid, BT), BT, 0, stream>>>(
        h2agg, b2, (long long)NB * hid, hid);

    // ---- pool + fc ----
    gat_pool<<<cdiv((long long)NB * hid, BT), BT, 0, stream>>>(h2agg, batch, pooled, Nn);
    gat_fc<<<cdiv(G * Cout, BT), BT, 0, stream>>>(pooled, Wfc, bfc, out, G, hid, Cout);
}
